// MultiHeadAttention_27023934227142
// MI455X (gfx1250) — compile-verified, matrix operand order checked by reading, not run on hardware
//
#include <hip/hip_runtime.h>
#include <hip/hip_bf16.h>

// ---------------------------------------------------------------------------
// MultiHeadAttention forward for MI455X (gfx1250, wave32, WMMA).
// B=2, S=2048, D=1024, H=16, HD=64.  Math in bf16 WMMA w/ f32 accumulation.
// ---------------------------------------------------------------------------

typedef __attribute__((ext_vector_type(16))) __bf16 v16bf;
typedef __attribute__((ext_vector_type(8)))  __bf16 v8bf;
typedef __attribute__((ext_vector_type(8)))  float  v8f;

static constexpr int BATCH  = 2;
static constexpr int SEQ    = 2048;
static constexpr int DMODEL = 1024;
static constexpr int NHEAD  = 16;
static constexpr int HDIM   = 64;

static __device__ __forceinline__ v8f wmma_bf16(v16bf a, v16bf b, v8f c) {
    // D = A(16x32 bf16) * B(32x16 bf16) + C(16x16 f32)
    return __builtin_amdgcn_wmma_f32_16x16x32_bf16(
        /*neg_a=*/false, a, /*neg_b=*/false, b,
        /*c_mod=*/(short)0, c, /*reuse_a=*/false, /*reuse_b=*/false);
}

static __device__ __forceinline__ v16bf pack16(v8bf lo, v8bf hi) {
    v16bf r;
#pragma unroll
    for (int i = 0; i < 8; ++i) { r[i] = lo[i]; r[i + 8] = hi[i]; }
    return r;
}

// ===========================================================================
// Kernel 1: fused QKV projection.  Y = X @ W + b, output bf16 to workspace.
// Block tile 128(M) x 64(N); 8 waves, each a 32x32 tile (2x2 WMMA frags).
// K stepped by 32.  A read from global fp32 (K-contiguous), B (weights)
// staged transposed into LDS as bf16 so B fragments are 2x ds_load_b128.
// grid = (M/128, N/64, 3) ; z picks q/k/v.
// ===========================================================================
__global__ __launch_bounds__(256)
void qkv_proj_kernel(const float* __restrict__ Xq, const float* __restrict__ Xk,
                     const float* __restrict__ Xv,
                     const float* __restrict__ Wq, const float* __restrict__ bq,
                     const float* __restrict__ Wk, const float* __restrict__ bk,
                     const float* __restrict__ Wv, const float* __restrict__ bv,
                     __bf16* __restrict__ Qo, __bf16* __restrict__ Ko,
                     __bf16* __restrict__ Vo)
{
    const int z = blockIdx.z;
    const float* X    = (z == 0) ? Xq : (z == 1) ? Xk : Xv;
    const float* W    = (z == 0) ? Wq : (z == 1) ? Wk : Wv;
    const float* bias = (z == 0) ? bq : (z == 1) ? bk : bv;
    __bf16*      Out  = (z == 0) ? Qo : (z == 1) ? Ko : Vo;

    const int K = DMODEL, N = DMODEL;
    const int tid  = threadIdx.x;
    const int w    = tid >> 5;
    const int lane = tid & 31;
    const int lm   = lane & 15;
    const int ko   = (lane & 16) ? 8 : 0;     // K half offset per ISA A/B layout
    const int wm   = w & 3;                   // 4 row groups of 32
    const int wn   = w >> 2;                  // 2 col groups of 32
    const int m0   = blockIdx.x * 128;
    const int n0   = blockIdx.y * 64;

    __shared__ __bf16 Bt[64][40];             // W^T tile: [n][k], padded stride

    v8f acc[2][2] = {};

    for (int kc = 0; kc < K; kc += 32) {
        __syncthreads();
        // Stage W[kc..kc+31, n0..n0+63] transposed -> Bt[n][k] (bf16), coalesced.
#pragma unroll
        for (int i = 0; i < 8; ++i) {
            int idx = tid + 256 * i;          // 0..2047
            int kk  = idx >> 6;               // 0..31
            int nn  = idx & 63;               // 0..63
            Bt[nn][kk] = (__bf16)W[(size_t)(kc + kk) * N + n0 + nn];
        }
        __syncthreads();

        v16bf afrag[2], bfrag[2];
#pragma unroll
        for (int s = 0; s < 2; ++s) {
            // A fragment: row = lane&15, K halves {ko..ko+7, ko+16..ko+23}
            const float* ap = X + (size_t)(m0 + wm * 32 + s * 16 + lm) * K + kc + ko;
            float4 x0 = *(const float4*)(ap);
            float4 x1 = *(const float4*)(ap + 4);
            float4 x2 = *(const float4*)(ap + 16);
            float4 x3 = *(const float4*)(ap + 20);
            v16bf a;
            a[0]=(__bf16)x0.x; a[1]=(__bf16)x0.y; a[2]=(__bf16)x0.z; a[3]=(__bf16)x0.w;
            a[4]=(__bf16)x1.x; a[5]=(__bf16)x1.y; a[6]=(__bf16)x1.z; a[7]=(__bf16)x1.w;
            a[8]=(__bf16)x2.x; a[9]=(__bf16)x2.y; a[10]=(__bf16)x2.z; a[11]=(__bf16)x2.w;
            a[12]=(__bf16)x3.x; a[13]=(__bf16)x3.y; a[14]=(__bf16)x3.z; a[15]=(__bf16)x3.w;
            afrag[s] = a;

            // B fragment from transposed LDS: col = lane&15, same K pattern.
            int nl = wn * 32 + s * 16 + lm;
            v8bf b0 = *(const v8bf*)&Bt[nl][ko];
            v8bf b1 = *(const v8bf*)&Bt[nl][ko + 16];
            bfrag[s] = pack16(b0, b1);
        }
#pragma unroll
        for (int i = 0; i < 2; ++i)
#pragma unroll
            for (int j = 0; j < 2; ++j)
                acc[i][j] = wmma_bf16(afrag[i], bfrag[j], acc[i][j]);
    }

    // Epilogue: + bias, store bf16.  C layout: row = r + (lane&16?8:0), col = lane&15.
#pragma unroll
    for (int i = 0; i < 2; ++i) {
#pragma unroll
        for (int j = 0; j < 2; ++j) {
            int col  = n0 + wn * 32 + j * 16 + lm;
            float bb = bias[col];
#pragma unroll
            for (int r = 0; r < 8; ++r) {
                int row = m0 + wm * 32 + i * 16 + r + ((lane & 16) ? 8 : 0);
                Out[(size_t)row * N + col] = (__bf16)(acc[i][j][r] + bb);
            }
        }
    }
}

// ===========================================================================
// Kernel 2: flash-style attention per (b, h, 64-query tile), key chunks of 256.
// ===========================================================================
static constexpr int CHUNK = 256;
static constexpr int SSTR  = 260;   // f32 scores row stride (padded)
static constexpr int PSTR  = 264;   // bf16 P / V^T row stride (padded, 16B-mult)

static constexpr size_t SM_P   = (size_t)64 * SSTR * 4;
static constexpr size_t SM_V   = SM_P + (size_t)64 * PSTR * 2;
static constexpr size_t SM_M   = SM_V + (size_t)64 * PSTR * 2;
static constexpr size_t SMEM_BYTES = SM_M + (3 * 64 + 2 * 256) * 4;

__global__ __launch_bounds__(256)
void attn_kernel(const __bf16* __restrict__ Qb, const __bf16* __restrict__ Kb,
                 const __bf16* __restrict__ Vb, __bf16* __restrict__ Ob)
{
    extern __shared__ char smem[];
    float*  scores = (float*)smem;                 // [64][SSTR]
    __bf16* Pb     = (__bf16*)(smem + SM_P);       // [64][PSTR]  exp(scores)
    __bf16* Vt     = (__bf16*)(smem + SM_V);       // [64(hd)][PSTR(key)]  V^T
    float*  mrow   = (float*)(smem + SM_M);        // running max
    float*  lrow   = mrow + 64;                    // running sum
    float*  arow   = lrow + 64;                    // rescale factor
    float*  pmax   = arow + 64;                    // [4][64] partial max
    float*  psum   = pmax + 256;                   // [4][64] partial sum

    const int tid  = threadIdx.x;
    const int w    = tid >> 5;
    const int lane = tid & 31;
    const int lm   = lane & 15;
    const int ko   = (lane & 16) ? 8 : 0;
    const int rg   = w & 3;                        // row group (16 rows)
    const int ch   = w >> 2;                       // col half / O col group
    const int q0   = blockIdx.x * 64;
    const int h    = blockIdx.y;
    const int b    = blockIdx.z;
    const size_t headoff = (size_t)h * HDIM;
    const size_t base    = (size_t)b * SEQ;

    if (tid < 64) { mrow[tid] = -1e30f; lrow[tid] = 0.f; }

    // Q fragments (reused across all key chunks): rows 16*rg .. +15, K=hd (2 steps).
    v16bf qa[2];
    {
        const __bf16* qp = Qb + (base + q0 + 16 * rg + lm) * DMODEL + headoff;
#pragma unroll
        for (int ks = 0; ks < 2; ++ks) {
            v8bf a0 = *(const v8bf*)(qp + 32 * ks + ko);
            v8bf a1 = *(const v8bf*)(qp + 32 * ks + ko + 16);
            qa[ks] = pack16(a0, a1);
        }
    }

    v8f oacc[2] = {};
    const int srow  = tid & 63;
    const int spart = tid >> 6;

    for (int kc = 0; kc < SEQ; kc += CHUNK) {
        __syncthreads();   // protect Vt/Pb of previous iteration

        // ---- stage V chunk transposed: Vt[hd][key] ----
#pragma unroll
        for (int i = 0; i < 8; ++i) {
            int v   = i * 256 + tid;               // 0..2047 (v8bf units)
            int key = v >> 3;                      // 0..255
            int hd8 = (v & 7) * 8;                 // 0,8,..,56
            const __bf16* vp = Vb + (base + kc + key) * DMODEL + headoff + hd8;
            v8bf vv = *(const v8bf*)vp;
#pragma unroll
            for (int e = 0; e < 8; ++e) Vt[(hd8 + e) * PSTR + key] = vv[e];
            if (kc + CHUNK < SEQ)
                __builtin_prefetch(vp + (size_t)CHUNK * DMODEL, 0, 1);
        }
        __syncthreads();

        // ---- S = Q K^T / sqrt(D): wave does rows 16*rg, cols 128*ch..+127 ----
#pragma unroll
        for (int t8 = 0; t8 < 8; ++t8) {
            int cb = 128 * ch + 16 * t8;
            // K^T fragment direct from global: col = key (lane&15), k = hd contiguous.
            const __bf16* kp = Kb + (base + kc + cb + lm) * DMODEL + headoff;
            v8f sacc = {};
#pragma unroll
            for (int ks = 0; ks < 2; ++ks) {
                v8bf b0 = *(const v8bf*)(kp + 32 * ks + ko);
                v8bf b1 = *(const v8bf*)(kp + 32 * ks + ko + 16);
                sacc = wmma_bf16(qa[ks], pack16(b0, b1), sacc);
            }
#pragma unroll
            for (int r = 0; r < 8; ++r) {
                int rr = 16 * rg + r + ((lane & 16) ? 8 : 0);
                scores[rr * SSTR + cb + lm] = sacc[r] * 0.03125f;  // 1/sqrt(1024)
            }
        }
        __syncthreads();

        // ---- online softmax: partial row max ----
        float mx = -1e30f;
#pragma unroll 8
        for (int c = 0; c < 64; ++c)
            mx = fmaxf(mx, scores[srow * SSTR + spart * 64 + c]);
        pmax[spart * 64 + srow] = mx;
        __syncthreads();
        if (tid < 64) {
            float mold = mrow[tid];
            float mc   = fmaxf(fmaxf(pmax[tid], pmax[64 + tid]),
                               fmaxf(pmax[128 + tid], pmax[192 + tid]));
            float mnew = fmaxf(mold, mc);
            float al   = __expf(mold - mnew);
            mrow[tid] = mnew;
            arow[tid] = al;
            lrow[tid] *= al;
        }
        __syncthreads();

        // ---- P = exp(S - m), bf16 into LDS; partial sums; rescale O ----
        {
            float mn  = mrow[srow];
            float sum = 0.f;
#pragma unroll 8
            for (int c = 0; c < 64; ++c) {
                float e = __expf(scores[srow * SSTR + spart * 64 + c] - mn);
                Pb[srow * PSTR + spart * 64 + c] = (__bf16)e;
                sum += e;
            }
            psum[spart * 64 + srow] = sum;
        }
#pragma unroll
        for (int j = 0; j < 2; ++j)
#pragma unroll
            for (int r = 0; r < 8; ++r)
                oacc[j][r] *= arow[16 * rg + r + ((lane & 16) ? 8 : 0)];
        __syncthreads();
        if (tid < 64)
            lrow[tid] += psum[tid] + psum[64 + tid] + psum[128 + tid] + psum[192 + tid];

        // ---- O += P * V : 8 K-steps of 32 over the 256-key chunk ----
#pragma unroll
        for (int ks = 0; ks < 8; ++ks) {
            const __bf16* pp = Pb + (16 * rg + lm) * PSTR + 32 * ks + ko;
            v16bf pa = pack16(*(const v8bf*)pp, *(const v8bf*)(pp + 16));
#pragma unroll
            for (int j = 0; j < 2; ++j) {
                int n = 32 * ch + 16 * j + lm;                 // hd column
                const __bf16* vp2 = Vt + n * PSTR + 32 * ks + ko;
                v16bf vb = pack16(*(const v8bf*)vp2, *(const v8bf*)(vp2 + 16));
                oacc[j] = wmma_bf16(pa, vb, oacc[j]);
            }
        }
    }
    __syncthreads();

    // ---- write O / l, merged-head layout [B*S, D] bf16 ----
#pragma unroll
    for (int j = 0; j < 2; ++j) {
        int col = 32 * ch + 16 * j + lm;
#pragma unroll
        for (int r = 0; r < 8; ++r) {
            int rr = 16 * rg + r + ((lane & 16) ? 8 : 0);
            float val = oacc[j][r] / lrow[rr];
            Ob[(base + q0 + rr) * DMODEL + headoff + col] = (__bf16)val;
        }
    }
}

// ===========================================================================
// Kernel 3: output projection.  out = A(bf16) @ Wo + bo, fp32 out.
// Same tiling as kernel 1; A fragments are direct 16B global loads.
// ===========================================================================
__global__ __launch_bounds__(256)
void out_proj_kernel(const __bf16* __restrict__ A, const float* __restrict__ W,
                     const float* __restrict__ bias, float* __restrict__ Out)
{
    const int K = DMODEL, N = DMODEL;
    const int tid  = threadIdx.x;
    const int w    = tid >> 5;
    const int lane = tid & 31;
    const int lm   = lane & 15;
    const int ko   = (lane & 16) ? 8 : 0;
    const int wm   = w & 3;
    const int wn   = w >> 2;
    const int m0   = blockIdx.x * 128;
    const int n0   = blockIdx.y * 64;

    __shared__ __bf16 Bt[64][40];

    v8f acc[2][2] = {};

    for (int kc = 0; kc < K; kc += 32) {
        __syncthreads();
#pragma unroll
        for (int i = 0; i < 8; ++i) {
            int idx = tid + 256 * i;
            int kk  = idx >> 6;
            int nn  = idx & 63;
            Bt[nn][kk] = (__bf16)W[(size_t)(kc + kk) * N + n0 + nn];
        }
        __syncthreads();

        v16bf afrag[2], bfrag[2];
#pragma unroll
        for (int s = 0; s < 2; ++s) {
            const __bf16* ap = A + (size_t)(m0 + wm * 32 + s * 16 + lm) * K + kc + ko;
            afrag[s] = pack16(*(const v8bf*)ap, *(const v8bf*)(ap + 16));
            int nl = wn * 32 + s * 16 + lm;
            bfrag[s] = pack16(*(const v8bf*)&Bt[nl][ko], *(const v8bf*)&Bt[nl][ko + 16]);
        }
#pragma unroll
        for (int i = 0; i < 2; ++i)
#pragma unroll
            for (int j = 0; j < 2; ++j)
                acc[i][j] = wmma_bf16(afrag[i], bfrag[j], acc[i][j]);
    }

#pragma unroll
    for (int i = 0; i < 2; ++i) {
#pragma unroll
        for (int j = 0; j < 2; ++j) {
            int col  = n0 + wn * 32 + j * 16 + lm;
            float bb = bias[col];
#pragma unroll
            for (int r = 0; r < 8; ++r) {
                int row = m0 + wm * 32 + i * 16 + r + ((lane & 16) ? 8 : 0);
                Out[(size_t)row * N + col] = acc[i][j][r] + bb;
            }
        }
    }
}

// ===========================================================================
extern "C" void kernel_launch(void* const* d_in, const int* in_sizes, int n_in,
                              void* d_out, int out_size, void* d_ws, size_t ws_size,
                              hipStream_t stream) {
    const float* Xq = (const float*)d_in[0];
    const float* Xk = (const float*)d_in[1];
    const float* Xv = (const float*)d_in[2];
    const float* Wq = (const float*)d_in[3];
    const float* bq = (const float*)d_in[4];
    const float* Wk = (const float*)d_in[5];
    const float* bk = (const float*)d_in[6];
    const float* Wv = (const float*)d_in[7];
    const float* bv = (const float*)d_in[8];
    const float* Wo = (const float*)d_in[9];
    const float* bo = (const float*)d_in[10];

    const size_t elems = (size_t)BATCH * SEQ * DMODEL;   // 4M elements
    __bf16* Qb = (__bf16*)d_ws;          // 8 MB
    __bf16* Kb = Qb + elems;             // 8 MB
    __bf16* Vb = Kb + elems;             // 8 MB
    __bf16* Ab = Vb + elems;             // 8 MB (pre-Wo attention output)

    // Allow >64KB dynamic LDS for the attention kernel (idempotent, deterministic).
    (void)hipFuncSetAttribute((const void*)attn_kernel,
                              hipFuncAttributeMaxDynamicSharedMemorySize,
                              (int)SMEM_BYTES);

    // 1) Q/K/V projections (z = 0/1/2), bf16 outputs to workspace.
    qkv_proj_kernel<<<dim3(32, 16, 3), 256, 0, stream>>>(
        Xq, Xk, Xv, Wq, bq, Wk, bk, Wv, bv, Qb, Kb, Vb);

    // 2) Attention: (64-query tiles, heads, batch).
    attn_kernel<<<dim3(SEQ / 64, NHEAD, BATCH), 256, SMEM_BYTES, stream>>>(
        Qb, Kb, Vb, Ab);

    // 3) Output projection to fp32.
    out_proj_kernel<<<dim3(32, 16), 256, 0, stream>>>(Ab, Wo, bo, (float*)d_out);
}
